// SineSynth_32985348833435
// MI455X (gfx1250) — compile-verified, matrix operand order checked by reading, not run on hardware
//
#include <hip/hip_runtime.h>
#include <hip/hip_bf16.h>
#include <math.h>

// SineSynth for MI455X (gfx1250, wave32).
//
// Pass 1 packs, per (b,s,control-frame j), everything the per-sample loop
// needs:  coef = {om_j, om_{j+1}-om_j, 16*(C[j-1]+C[j]), amp_j}  (float4)
//         dam  = amp_{j+1} - amp_j                               (float)
// where C is the prefix sum of control-point omegas.  Closed form of the
// cumsum of the linearly-upsampled omega at sample t = 16 + 32j + k:
//   phase = 16*(C[j-1]+C[j]) + (k+1)*om_j + ((k+1)^2/64)*(om_{j+1}-om_j)
// (head t<16: phase=(t+1)*om_0; tail j=511 degenerates correctly).
//
// Pass 2: each wave32 owns 16 output samples; the 256-sine reduction is a
// ones(16x32) x P(32x16) column-sum done with V_WMMA_F32_16X16X32_F16,
// accumulated in f32 across 8 chunks.  No LDS / cross-lane shuffles.

typedef __attribute__((ext_vector_type(16))) _Float16 v16h;
typedef __attribute__((ext_vector_type(8)))  float    v8f;

#define SINES   256
#define CTRL    512
#define T_OUT   16384           // 512 * 32
#define NBATCH  4
#define ROWS    (NBATCH * SINES)

__device__ __constant__ float kTwoPi    = 6.28318530717958647692f;
__device__ __constant__ float kInvTwoPi = 0.15915494309189533577f;
__device__ __constant__ float kOmScale  = 6.28318530717958647692f / 44100.0f; // 2*pi/FS

// ---------------------------------------------------------------------------
// Pass 1 (packed): per-row serial scan, emits float4 coef + float dam.
// 1024 rows x 512 steps: trivial, L2-resident.
// ---------------------------------------------------------------------------
__global__ __launch_bounds__(256)
void sinesynth_pack_kernel(const float* __restrict__ freq,
                           const float* __restrict__ amp,
                           float4* __restrict__ coef,
                           float*  __restrict__ dam)
{
    int r = blockIdx.x * blockDim.x + threadIdx.x;
    if (r >= ROWS) return;
    const float* f = freq + (size_t)r * CTRL;
    const float* a = amp  + (size_t)r * CTRL;
    float4* c4 = coef + (size_t)r * CTRL;
    float*  dm = dam  + (size_t)r * CTRL;

    float cprev = 0.0f, ccur = 0.0f;
    for (int j = 0; j < CTRL; ++j) {
        const int   i1  = (j + 1 < CTRL) ? (j + 1) : (CTRL - 1);
        const float omj = f[j]  * kOmScale;
        const float omi = f[i1] * kOmScale;
        ccur += omj;                                  // C[j]
        const float aj = a[j], ai = a[i1];
        c4[j] = make_float4(omj, omi - omj, 16.0f * (cprev + ccur), aj);
        dm[j] = ai - aj;
        cprev = ccur;
    }
}

// ---------------------------------------------------------------------------
// Pass 1 (fallback): shifted prefix row of 513 floats (leading 0) so the
// main loop reads {C[j-1], C[j]} branch-free.  ~2.1 MB workspace.
// ---------------------------------------------------------------------------
__global__ __launch_bounds__(256)
void sinesynth_prefix_kernel(const float* __restrict__ freq,
                             float* __restrict__ Cs)
{
    int r = blockIdx.x * blockDim.x + threadIdx.x;
    if (r >= ROWS) return;
    const float* f = freq + (size_t)r * CTRL;
    float*       c = Cs   + (size_t)r * (CTRL + 1);
    float acc = 0.0f;
    c[0] = 0.0f;
    for (int i = 0; i < CTRL; ++i) {
        acc += f[i] * kOmScale;
        c[i + 1] = acc;
    }
}

// ---------------------------------------------------------------------------
// Shared per-lane time decomposition.
// ---------------------------------------------------------------------------
struct TimeParams {
    int   j;
    bool  head;
    float w, kp1, quad;
};

__device__ __forceinline__ TimeParams time_params(int t)
{
    TimeParams tp;
    tp.head = (t < 16);
    const int u = tp.head ? 0 : (t - 16);
    tp.j = u >> 5;
    const int k = u & 31;
    tp.w    = tp.head ? 0.0f : ((float)k + 0.5f) * (1.0f / 32.0f);
    tp.kp1  = tp.head ? (float)(t + 1) : (float)(k + 1);
    tp.quad = tp.head ? 0.0f : (tp.kp1 * tp.kp1) * (1.0f / 64.0f);
    return tp;
}

__device__ __forceinline__ float wrap_sin(float ph)
{
    // Python float %: x - floor(x/y)*y (ph >= 0 here), then sin.
    ph = ph - kTwoPi * floorf(ph * kInvTwoPi);
    return __sinf(ph);
}

// ---------------------------------------------------------------------------
// Pass 2 (packed coefficients): 2 loads (b128 + b32) per product.
// ---------------------------------------------------------------------------
__global__ __launch_bounds__(256)
void sinesynth_main_packed(const float4* __restrict__ coef,
                           const float*  __restrict__ dam,
                           float* __restrict__ out)
{
    const int lane  = threadIdx.x & 31;
    const int wave  = threadIdx.x >> 5;               // wave32: 8 waves / block
    const int tilesPerBatch = T_OUT / 128;            // 128
    const int b     = blockIdx.x / tilesPerBatch;
    const int t     = (blockIdx.x % tilesPerBatch) * 128 + wave * 16 + (lane & 15);
    const int sHalf = (lane >> 4) << 4;               // K half: 0 or 16

    const TimeParams tp = time_params(t);
    const size_t bBase = (size_t)b * SINES * CTRL + tp.j;

    v16h ones;
    #pragma unroll
    for (int e = 0; e < 16; ++e) ones[e] = (_Float16)1.0f;

    v8f acc = {};

    for (int sc = 0; sc < SINES; sc += 32) {
        if (sc + 32 < SINES) {
            // gfx1250 global_prefetch_b8: pull next chunk's coefficient lines.
            __builtin_prefetch(coef + bBase + (size_t)(sc + 32 + sHalf) * CTRL, 0, 0);
        }
        v16h bm;
        #pragma unroll
        for (int e = 0; e < 16; ++e) {
            const size_t idx = bBase + (size_t)(sc + sHalf + e) * CTRL;
            const float4 c4  = coef[idx];
            const float  da  = dam[idx];
            const float base = tp.head ? 0.0f : c4.z;
            const float ph   = base + tp.kp1 * c4.x + tp.quad * c4.y;
            const float av   = c4.w + tp.w * da;
            bm[e] = (_Float16)(av * wrap_sin(ph));
        }
        acc = __builtin_amdgcn_wmma_f32_16x16x32_f16(
            false, ones, false, bm, (short)0, acc, false, false);
    }

    // D VGPR0: lanes 0-15 hold M=0, N=lane -> the 16 column sums.
    if ((lane & 16) == 0) {
        out[(size_t)b * T_OUT + t] = acc[0];
    }
}

// ---------------------------------------------------------------------------
// Pass 2 (fallback, shifted-prefix workspace): 6 b32 loads, branch-free.
// ---------------------------------------------------------------------------
__global__ __launch_bounds__(256)
void sinesynth_main_shift(const float* __restrict__ freq,
                          const float* __restrict__ amp,
                          const float* __restrict__ Cs,
                          float* __restrict__ out)
{
    const int lane  = threadIdx.x & 31;
    const int wave  = threadIdx.x >> 5;
    const int tilesPerBatch = T_OUT / 128;
    const int b     = blockIdx.x / tilesPerBatch;
    const int t     = (blockIdx.x % tilesPerBatch) * 128 + wave * 16 + (lane & 15);
    const int sHalf = (lane >> 4) << 4;

    const TimeParams tp = time_params(t);
    const int i1 = (tp.j + 1 < CTRL) ? (tp.j + 1) : (CTRL - 1);
    const size_t bRow = (size_t)b * SINES;

    v16h ones;
    #pragma unroll
    for (int e = 0; e < 16; ++e) ones[e] = (_Float16)1.0f;

    v8f acc = {};

    for (int sc = 0; sc < SINES; sc += 32) {
        v16h bm;
        #pragma unroll
        for (int e = 0; e < 16; ++e) {
            const size_t r    = bRow + (size_t)(sc + sHalf + e);
            const size_t row  = r * CTRL;
            const size_t rowC = r * (CTRL + 1);
            const float omj = freq[row + tp.j] * kOmScale;
            const float omi = freq[row + i1]   * kOmScale;
            const float cm1 = Cs[rowC + tp.j];          // C[j-1] (0 for j==0)
            const float cj  = Cs[rowC + tp.j + 1];      // C[j]
            const float base = tp.head ? 0.0f : 16.0f * (cm1 + cj);
            const float ph   = base + tp.kp1 * omj + tp.quad * (omi - omj);
            const float aj = amp[row + tp.j];
            const float ai = amp[row + i1];
            const float av = aj + tp.w * (ai - aj);
            bm[e] = (_Float16)(av * wrap_sin(ph));
        }
        acc = __builtin_amdgcn_wmma_f32_16x16x32_f16(
            false, ones, false, bm, (short)0, acc, false, false);
    }

    if ((lane & 16) == 0) {
        out[(size_t)b * T_OUT + t] = acc[0];
    }
}

extern "C" void kernel_launch(void* const* d_in, const int* in_sizes, int n_in,
                              void* d_out, int out_size, void* d_ws, size_t ws_size,
                              hipStream_t stream)
{
    const float* freq = (const float*)d_in[0];   // [4,256,512] f32
    const float* amp  = (const float*)d_in[1];   // [4,256,512] f32
    float*       out  = (float*)d_out;           // [4,1,16384] f32

    (void)in_sizes; (void)n_in; (void)out_size;

    const size_t nElem       = (size_t)ROWS * CTRL;
    const size_t packedBytes = nElem * sizeof(float4) + nElem * sizeof(float); // 10 MB
    const size_t shiftBytes  = (size_t)ROWS * (CTRL + 1) * sizeof(float);      // 2.1 MB

    const dim3 packGrid((ROWS + 255) / 256), packBlk(256);
    const dim3 mainGrid(NBATCH * (T_OUT / 128)), mainBlk(256);   // 512 x 256

    if (ws_size >= packedBytes) {
        float4* coef = (float4*)d_ws;
        float*  dam  = (float*)(coef + nElem);
        sinesynth_pack_kernel<<<packGrid, packBlk, 0, stream>>>(freq, amp, coef, dam);
        sinesynth_main_packed<<<mainGrid, mainBlk, 0, stream>>>(coef, dam, out);
    } else if (ws_size >= shiftBytes) {
        float* Cs = (float*)d_ws;
        sinesynth_prefix_kernel<<<packGrid, packBlk, 0, stream>>>(freq, Cs);
        sinesynth_main_shift<<<mainGrid, mainBlk, 0, stream>>>(freq, amp, Cs, out);
    }
    // (ws_size below 2.1 MB is not expected for this problem size.)
}
